// GLSGRv3FeaturePyramidNetwork_24910810316995
// MI455X (gfx1250) — matrix operand order repaired and checked by reading, not compiled, not run on hardware
//
#include <hip/hip_runtime.h>
#include <math.h>

// ---------------- constants ----------------
#define NNODES 3600
#define KNN    7
#define E7     (NNODES * KNN)   // 25200 top-k edges; self loops are edges [E7, E7+NNODES)
#define ETOT   (E7 + NNODES)    // 28800

typedef __attribute__((ext_vector_type(16))) _Float16 v16h;
typedef __attribute__((ext_vector_type(8)))  float    v8f;

// =====================================================================
// Wave-level WMMA GEMM, register-blocked 16x64 per wave (4 accumulators).
//   C[M,N] = epilogue( A[M,K] * B + bias )   (+ add)
//   TB==0: B is row-major [K,N]   (x @ W, Linear layout)
//   TB==1: B is row-major [N,K]   (C = A * B^T, e.g. emb @ emb^T)
// M multiple of 16, N multiple of 16 (remainder group handled uniformly),
// K multiple of 32. relu_act: 1 -> ReLU.
// =====================================================================
template <int TB>
__global__ void gemm_wmma(const float* __restrict__ A, const float* __restrict__ B,
                          const float* __restrict__ bias, const float* __restrict__ addp,
                          float* __restrict__ C, int M, int N, int K, int relu_act) {
  int wave = blockIdx.x * (blockDim.x >> 5) + (threadIdx.x >> 5);
  int lane = threadIdx.x & 31;
  int nt = N >> 4;                 // 16-col tiles
  int ng = (nt + 3) >> 2;          // groups of 4 tiles
  int total = (M >> 4) * ng;
  if (wave >= total) return;       // wave-uniform: EXEC stays all-ones for WMMA
  int m0 = (wave / ng) << 4;
  int g  = wave % ng;
  int n0 = g << 6;                 // group base column
  int rem = nt - (g << 2);
  int nsub = rem < 4 ? rem : 4;    // wave-uniform sub-tile count
  int r  = lane & 15;
  int hi = lane >> 4;              // half-wave select
  const float* Arow = A + (size_t)(m0 + r) * K;
  v8f acc[4] = {};
  for (int kb = 0; kb < K; kb += 32) {
    v16h a;
    int ka = kb + hi * 16;         // A: hi=0 -> K 0..15 ; hi=1 -> 16..31
#pragma unroll
    for (int j = 0; j < 8; ++j) {
      int koff = ka + 2 * j;
      a[2 * j]     = (_Float16)Arow[koff];
      a[2 * j + 1] = (_Float16)Arow[koff + 1];
    }
#pragma unroll
    for (int s = 0; s < 4; ++s) {
      if (s < nsub) {              // uniform guard
        v16h b;
        if (TB == 0) {
          const float* Bc = B + (size_t)(kb + hi * 16) * N + (n0 + 16 * s + r);
#pragma unroll
          for (int j = 0; j < 8; ++j) {      // coalesced across lanes at each k-row
            b[2 * j]     = (_Float16)Bc[(size_t)(2 * j) * N];
            b[2 * j + 1] = (_Float16)Bc[(size_t)(2 * j + 1) * N];
          }
        } else {
          const float* Br = B + (size_t)(n0 + 16 * s + r) * K + kb + hi * 16;
#pragma unroll
          for (int j = 0; j < 8; ++j) {      // contiguous 16 floats per lane
            b[2 * j]     = (_Float16)Br[2 * j];
            b[2 * j + 1] = (_Float16)Br[2 * j + 1];
          }
        }
        acc[s] = __builtin_amdgcn_wmma_f32_16x16x32_f16(false, a, false, b, (short)0,
                                                        acc[s], false, false);
      }
    }
  }
  int mbase = m0 + hi * 8;         // C/D layout: lanes 16-31 hold rows M+8..M+15
#pragma unroll
  for (int s = 0; s < 4; ++s) {
    if (s < nsub) {
      int n = n0 + 16 * s + r;
      float bv = bias ? bias[n] : 0.0f;
#pragma unroll
      for (int q = 0; q < 8; ++q) {
        float v = acc[s][q] + bv;
        if (relu_act) v = v > 0.0f ? v : 0.0f;
        if (addp) v += addp[(size_t)(mbase + q) * N + n];
        C[(size_t)(mbase + q) * N + n] = v;
      }
    }
  }
}

// =====================================================================
// Patch-embed GEMM: t1[3600,256] = ReLU( patches[3600,6400] * W1^T + b1 )
// A gathered (im2col on the fly) from x[256,300,300]; gather amortized over
// 4 N-tiles per wave. B=conv1_w [256,6400] (NT form).
// =====================================================================
__global__ void gemm_patch(const float* __restrict__ x, const float* __restrict__ W,
                           const float* __restrict__ bias, float* __restrict__ C) {
  const int N = 256, K = 6400;
  int wave = blockIdx.x * (blockDim.x >> 5) + (threadIdx.x >> 5);
  int lane = threadIdx.x & 31;
  int ng = (N >> 4) >> 2;                     // 4 groups of 64 cols (exact)
  if (wave >= (NNODES >> 4) * ng) return;
  int m0 = (wave / ng) << 4;
  int n0 = (wave % ng) << 6;
  int r = lane & 15, hi = lane >> 4;
  int m = m0 + r;
  int pi = (m / 60) * 5, pj = (m % 60) * 5;
  const float* base = x + (size_t)pi * 300 + pj;
  v8f acc[4] = {};
  for (int kb = 0; kb < K; kb += 32) {
    v16h a;
    int ka = kb + hi * 16;
#pragma unroll
    for (int j = 0; j < 8; ++j) {
#pragma unroll
      for (int t2 = 0; t2 < 2; ++t2) {
        int k = ka + 2 * j + t2;
        int c = k / 25, remk = k - c * 25;
        int rr = remk / 5, cc = remk - rr * 5;
        a[2 * j + t2] = (_Float16)base[(size_t)c * 90000 + rr * 300 + cc];
      }
    }
#pragma unroll
    for (int s = 0; s < 4; ++s) {
      v16h b;
      const float* Br = W + (size_t)(n0 + 16 * s + r) * K + kb + hi * 16;
#pragma unroll
      for (int j = 0; j < 8; ++j) {
        b[2 * j]     = (_Float16)Br[2 * j];
        b[2 * j + 1] = (_Float16)Br[2 * j + 1];
      }
      acc[s] = __builtin_amdgcn_wmma_f32_16x16x32_f16(false, a, false, b, (short)0,
                                                      acc[s], false, false);
    }
  }
  int mbase = m0 + hi * 8;
#pragma unroll
  for (int s = 0; s < 4; ++s) {
    int n = n0 + 16 * s + r;
    float bv = bias[n];
#pragma unroll
    for (int q = 0; q < 8; ++q) {
      float v = acc[s][q] + bv;
      C[(size_t)(mbase + q) * N + n] = v > 0.0f ? v : 0.0f;
    }
  }
}

// =====================================================================
// Top-7 of each adjacency row (largest value, tie -> lowest index).
// edge e = 7*i + t : src=idx, dst=i, mask=(i < idx) && (val != 0)
// =====================================================================
__global__ void topk_feature(const float* __restrict__ adj, int* __restrict__ srcA,
                             unsigned char* __restrict__ maskA) {
  __shared__ float row[NNODES];
  __shared__ float sv[256];
  __shared__ int   si[256];
  int i = blockIdx.x, tid = threadIdx.x;
  for (int j = tid; j < NNODES; j += 256) row[j] = adj[(size_t)i * NNODES + j];
  __syncthreads();
  for (int t = 0; t < KNN; ++t) {
    float best = -3.4e38f; int bidx = NNODES;
    for (int j = tid; j < NNODES; j += 256) {
      float v = row[j];
      if (v > best) { best = v; bidx = j; }
    }
    sv[tid] = best; si[tid] = bidx; __syncthreads();
    for (int s = 128; s > 0; s >>= 1) {
      if (tid < s) {
        if (sv[tid + s] > sv[tid] || (sv[tid + s] == sv[tid] && si[tid + s] < si[tid])) {
          sv[tid] = sv[tid + s]; si[tid] = si[tid + s];
        }
      }
      __syncthreads();
    }
    if (tid == 0) {
      int e = i * KNN + t;
      srcA[e]  = si[0];
      maskA[e] = (i < si[0] && sv[0] != 0.0f) ? 1 : 0;
      row[si[0]] = -3.4e38f;
    }
    __syncthreads();
  }
}

// =====================================================================
// Spatial kNN: 7 smallest distances (tie -> lowest index), reference formula
// dist = sqrt(max(|ci|^2+|cj|^2-2 ci.cj, 0)); mask=(i <= idx) && (dist != 0)
// =====================================================================
__global__ void topk_spatial(int* __restrict__ srcA, unsigned char* __restrict__ maskA) {
  __shared__ float row[NNODES];
  __shared__ float sv[256];
  __shared__ int   si[256];
  int i = blockIdx.x, tid = threadIdx.x;
  float yi = 5.0f * (float)(i / 60), xi = 5.0f * (float)(i % 60);
  float sqi = yi * yi + xi * xi;
  for (int j = tid; j < NNODES; j += 256) {
    float yj = 5.0f * (float)(j / 60), xj = 5.0f * (float)(j % 60);
    float sqj = yj * yj + xj * xj;
    float d2 = sqi + sqj - 2.0f * (yi * yj + xi * xj);
    row[j] = sqrtf(fmaxf(d2, 0.0f));
  }
  __syncthreads();
  for (int t = 0; t < KNN; ++t) {
    float best = 3.4e38f; int bidx = NNODES;
    for (int j = tid; j < NNODES; j += 256) {
      float v = row[j];
      if (v < best) { best = v; bidx = j; }
    }
    sv[tid] = best; si[tid] = bidx; __syncthreads();
    for (int s = 128; s > 0; s >>= 1) {
      if (tid < s) {
        if (sv[tid + s] < sv[tid] || (sv[tid + s] == sv[tid] && si[tid + s] < si[tid])) {
          sv[tid] = sv[tid + s]; si[tid] = si[tid + s];
        }
      }
      __syncthreads();
    }
    if (tid == 0) {
      int e = i * KNN + t;
      srcA[e]  = si[0];
      maskA[e] = (i <= si[0] && sv[0] != 0.0f) ? 1 : 0;
      row[si[0]] = 3.4e38f;
    }
    __syncthreads();
  }
}

// =====================================================================
// GATv2 edge logits: logit[e,h] = mask ? att[h].leaky(xl[src,h,:]+xr[dst,h,:]) : -1e30
// grid = ETOT blocks, blockDim = 32*H (one wave per head).
// =====================================================================
__global__ void edge_logits(const float* __restrict__ xl, const float* __restrict__ xr,
                            const float* __restrict__ att, const int* __restrict__ srcA,
                            const unsigned char* __restrict__ maskA,
                            float* __restrict__ logit, int H, int F) {
  int e = blockIdx.x;
  int h = threadIdx.x >> 5, lane = threadIdx.x & 31;
  int dstn = (e < E7) ? (e / KNN) : (e - E7);
  int s    = (e < E7) ? srcA[e] : dstn;
  bool mk  = (e < E7) ? (maskA[e] != 0) : true;
  const float* pl = xl + (size_t)s * H * F + h * F;
  const float* pr = xr + (size_t)dstn * H * F + h * F;
  const float* pa = att + h * F;
  float acc = 0.0f;
  for (int f = lane; f < F; f += 32) {
    float v = pl[f] + pr[f];
    v = v > 0.0f ? v : 0.2f * v;
    acc += v * pa[f];
  }
  for (int off = 16; off > 0; off >>= 1) acc += __shfl_down(acc, off, 32);
  if (lane == 0) logit[e * H + h] = mk ? acc : -1e30f;
}

// =====================================================================
// Per-node segment softmax + aggregation + bias + ELU.
// Node i's incoming edges: {7i..7i+6} plus self loop E7+i.
// =====================================================================
__global__ void gat_aggregate(const float* __restrict__ xl, const float* __restrict__ logit,
                              const int* __restrict__ srcA, const unsigned char* __restrict__ maskA,
                              const float* __restrict__ bias, float* __restrict__ out,
                              int H, int F) {
  __shared__ float s_a[8 * 8];
  __shared__ int   s_src[8];
  int i = blockIdx.x, tid = threadIdx.x;
  int HF = H * F;
  if (tid < 8) s_src[tid] = (tid < KNN) ? srcA[i * KNN + tid] : i;
  if (tid < H) {
    int h = tid;
    float lg[8];
    float m = -3.4e38f;
    for (int j = 0; j < KNN; ++j) { lg[j] = logit[(i * KNN + j) * H + h]; m = fmaxf(m, lg[j]); }
    lg[7] = logit[(E7 + i) * H + h];
    m = fmaxf(m, lg[7]);
    float denom = 0.0f, ex[8];
    for (int j = 0; j < 8; ++j) {
      bool mk = (j == KNN) ? true : (maskA[i * KNN + j] != 0);
      float v = mk ? __expf(lg[j] - m) : 0.0f;
      ex[j] = v; denom += v;
    }
    float inv = 1.0f / denom;
    for (int j = 0; j < 8; ++j) s_a[j * H + h] = ex[j] * inv;
  }
  __syncthreads();
  for (int o = tid; o < HF; o += blockDim.x) {
    int h = o / F;
    float acc = 0.0f;
    for (int j = 0; j < 8; ++j)
      acc += s_a[j * H + h] * xl[(size_t)s_src[j] * HF + o];
    float v = acc + bias[o];
    out[(size_t)i * HF + o] = v > 0.0f ? v : (__expf(v) - 1.0f);   // ELU
  }
}

// cat[i, 0:256] = a[i], cat[i, 256:512] = b[i]
__global__ void concat_rows(const float* __restrict__ a, const float* __restrict__ b,
                            float* __restrict__ out) {
  int idx = blockIdx.x * blockDim.x + threadIdx.x;
  if (idx >= NNODES * 512) return;
  int i = idx >> 9, c = idx & 511;
  out[idx] = (c < 256) ? a[i * 256 + c] : b[i * 256 + (c - 256)];
}

// ---------------------------------------------------------------------
static inline int gemm_grid(int M, int N) {
  int nt = N >> 4;
  int ng = (nt + 3) >> 2;        // 4 tiles per wave along N
  int t = (M >> 4) * ng;
  return (t + 7) / 8;            // 8 waves (256 threads) per block
}

extern "C" void kernel_launch(void* const* d_in, const int* in_sizes, int n_in,
                              void* d_out, int out_size, void* d_ws, size_t ws_size,
                              hipStream_t stream) {
  const float* x       = (const float*)d_in[0];
  const float* conv1_w = (const float*)d_in[1];
  const float* conv1_b = (const float*)d_in[2];
  const float* fc2_w   = (const float*)d_in[3];
  const float* fc2_b   = (const float*)d_in[4];
  const float* fc3_w   = (const float*)d_in[5];
  const float* fc3_b   = (const float*)d_in[6];
  const float *g_wl[4], *g_bl[4], *g_wr[4], *g_br[4], *g_att[4], *g_bias[4];
  for (int g = 0; g < 4; ++g) {
    int b = 7 + 6 * g;
    g_wl[g]   = (const float*)d_in[b + 0];
    g_bl[g]   = (const float*)d_in[b + 1];
    g_wr[g]   = (const float*)d_in[b + 2];
    g_br[g]   = (const float*)d_in[b + 3];
    g_att[g]  = (const float*)d_in[b + 4];
    g_bias[g] = (const float*)d_in[b + 5];
  }

  // ---- workspace layout (floats) ----
  float* ws = (float*)d_ws;
  const size_t SMALL = (size_t)NNODES * 256;         //   921,600
  const size_t BIG   = (size_t)NNODES * 2048;        // 7,372,800
  float* t1    = ws;                                  // conv out / xl_small
  float* emb   = t1 + SMALL;
  float* h_g2  = emb + SMALL;
  float* h_g4  = h_g2 + SMALL;
  float* xr_s  = h_g4 + SMALL;
  float* cat   = xr_s + SMALL;                        // [3600,512]
  float* logit = cat + (size_t)NNODES * 512;          // ETOT*8 = 230,400
  float* big0  = logit + (size_t)ETOT * 8;            // xl big; adj overlays big0..big1
  float* big1  = big0 + BIG;                          // xr big
  float* big2  = big1 + BIG;                          // GAT big output
  float* adj   = big0;                                // 12.96M floats, consumed before big0/big1 reused
  char* tail   = (char*)(big2 + BIG);
  int* src1 = (int*)tail;
  int* src2 = src1 + E7;
  unsigned char* mask1 = (unsigned char*)(src2 + E7);
  unsigned char* mask2 = mask1 + E7;

  // ---- 1. patch embed + fc2 ----
  gemm_patch<<<gemm_grid(NNODES, 256), 256, 0, stream>>>(x, conv1_w, conv1_b, t1);
  gemm_wmma<0><<<gemm_grid(NNODES, 256), 256, 0, stream>>>(t1, fc2_w, fc2_b, nullptr, emb,
                                                           NNODES, 256, 256, 1);
  // ---- 2. graphs ----
  gemm_wmma<1><<<gemm_grid(NNODES, NNODES), 256, 0, stream>>>(emb, emb, nullptr, nullptr, adj,
                                                              NNODES, NNODES, 256, 0);
  topk_feature<<<NNODES, 256, 0, stream>>>(adj, src1, mask1);
  topk_spatial<<<NNODES, 256, 0, stream>>>(src2, mask2);

  // ---- 3. GAT layer 1 (feature graph, H=8) ----
  gemm_wmma<0><<<gemm_grid(NNODES, 2048), 256, 0, stream>>>(emb, g_wl[0], g_bl[0], nullptr, big0,
                                                            NNODES, 2048, 256, 0);
  gemm_wmma<0><<<gemm_grid(NNODES, 2048), 256, 0, stream>>>(emb, g_wr[0], g_br[0], nullptr, big1,
                                                            NNODES, 2048, 256, 0);
  edge_logits<<<ETOT, 256, 0, stream>>>(big0, big1, g_att[0], src1, mask1, logit, 8, 256);
  gat_aggregate<<<NNODES, 256, 0, stream>>>(big0, logit, src1, mask1, g_bias[0], big2, 8, 256);

  // ---- 4. GAT layer 2 (feature graph, H=1) ----
  gemm_wmma<0><<<gemm_grid(NNODES, 256), 256, 0, stream>>>(big2, g_wl[1], g_bl[1], nullptr, t1,
                                                           NNODES, 256, 2048, 0);
  gemm_wmma<0><<<gemm_grid(NNODES, 256), 256, 0, stream>>>(big2, g_wr[1], g_br[1], nullptr, xr_s,
                                                           NNODES, 256, 2048, 0);
  edge_logits<<<ETOT, 32, 0, stream>>>(t1, xr_s, g_att[1], src1, mask1, logit, 1, 256);
  gat_aggregate<<<NNODES, 256, 0, stream>>>(t1, logit, src1, mask1, g_bias[1], h_g2, 1, 256);

  // ---- 5. GAT layer 3 (spatial graph, H=8) ----
  gemm_wmma<0><<<gemm_grid(NNODES, 2048), 256, 0, stream>>>(emb, g_wl[2], g_bl[2], nullptr, big0,
                                                            NNODES, 2048, 256, 0);
  gemm_wmma<0><<<gemm_grid(NNODES, 2048), 256, 0, stream>>>(emb, g_wr[2], g_br[2], nullptr, big1,
                                                            NNODES, 2048, 256, 0);
  edge_logits<<<ETOT, 256, 0, stream>>>(big0, big1, g_att[2], src2, mask2, logit, 8, 256);
  gat_aggregate<<<NNODES, 256, 0, stream>>>(big0, logit, src2, mask2, g_bias[2], big2, 8, 256);

  // ---- 6. GAT layer 4 (spatial graph, H=1) ----
  gemm_wmma<0><<<gemm_grid(NNODES, 256), 256, 0, stream>>>(big2, g_wl[3], g_bl[3], nullptr, t1,
                                                           NNODES, 256, 2048, 0);
  gemm_wmma<0><<<gemm_grid(NNODES, 256), 256, 0, stream>>>(big2, g_wr[3], g_br[3], nullptr, xr_s,
                                                           NNODES, 256, 2048, 0);
  edge_logits<<<ETOT, 32, 0, stream>>>(t1, xr_s, g_att[3], src2, mask2, logit, 1, 256);
  gat_aggregate<<<NNODES, 256, 0, stream>>>(t1, logit, src2, mask2, g_bias[3], h_g4, 1, 256);

  // ---- 7. fc3 with fused ReLU + residual add of h_g4 ----
  concat_rows<<<(NNODES * 512 + 255) / 256, 256, 0, stream>>>(h_g2, h_g4, cat);
  gemm_wmma<0><<<gemm_grid(NNODES, 256), 256, 0, stream>>>(cat, fc3_w, fc3_b, h_g4,
                                                           (float*)d_out, NNODES, 256, 512, 1);
}
